// LDEModule_67860483277134
// MI455X (gfx1250) — matrix operand order repaired and checked by reading, not compiled, not run on hardware
//
#include <hip/hip_runtime.h>

// ---------------------------------------------------------------------------
// Fused Levy-SDE kernel for MI455X (gfx1250, wave32, WMMA).
//   h0   = x_last @ W_down + b_down                           (WMMA f16, K=128)
//   gate / diff_micro / diff_scale                            (VALU, prologue)
//   25x:  h += relu(h @ W_drift + b_drift)*dt + ds*coef*eta   (WMMA f16, K=512)
// Rows are independent -> each block owns 64 rows for all 25 steps.
// h master lives in fp32 registers in WMMA C-layout; an f16 shadow in LDS is
// the A operand. W is pre-converted once to f16 in B-fragment dword layout
// (in d_ws) and staged per K=32 chunk with GLOBAL_LOAD_ASYNC_TO_LDS_B128
// (double-buffered, ASYNCcnt-tracked). Fallbacks: plain b128 copy if the
// async builtin is absent; in-kernel f32->f16 staging if d_ws is too small.
// ---------------------------------------------------------------------------

typedef _Float16     v16h __attribute__((ext_vector_type(16)));
typedef float        v8f  __attribute__((ext_vector_type(8)));
typedef unsigned int v4u  __attribute__((ext_vector_type(4)));
typedef int          v4i  __attribute__((ext_vector_type(4)));

union ABFrag { v16h h; v4u u[2]; };

#define B_     4096
#define T_     30
#define DIN    128
#define DH     512
#define MID_   100
#define STEPS_ 25

#define ROWS      64    // rows per block
#define HA_STRIDE 520   // f16 per hA row (512 + 8 pad -> 260 dwords, bank-spread)
#define WB_STRIDE 20    // dwords per column in W staging buffer (16B aligned)
#define CH_DW     8192  // dwords per staged K=32 chunk (512 cols * 16 k2)
#define N_CHUNKS  20    // 4 (W_down) + 16 (W_drift)

#if __has_builtin(__builtin_amdgcn_global_load_async_to_lds_b128)
#define HAVE_ASYNC_LDS 1
#else
#define HAVE_ASYNC_LDS 0
#endif

__device__ __forceinline__ void wait_async0() {
#if __has_builtin(__builtin_amdgcn_s_wait_asynccnt)
    __builtin_amdgcn_s_wait_asynccnt(0);
#else
    asm volatile("s_wait_asynccnt 0x0" ::: "memory");
#endif
}

#if HAVE_ASYNC_LDS
__device__ __forceinline__ void async_cp16(const unsigned int* gsrc, unsigned int* ldst) {
    __builtin_amdgcn_global_load_async_to_lds_b128(
        (__attribute__((address_space(1))) v4i*)gsrc,
        (__attribute__((address_space(3))) v4i*)ldst,
        0, 0);
}
#endif

// One-time: convert W_down (chunks 0..3) + W_drift (chunks 4..19) from f32
// [K][512] row-major into f16 packed dwords, chunk-contiguous B-frag layout:
//   dst[c*8192 + n*16 + k2] = pack_f16(W[kc*32+2k2][n], W[kc*32+2k2+1][n])
__global__ __launch_bounds__(256) void prep_w(const float* __restrict__ W_down,
                                              const float* __restrict__ W_drift,
                                              unsigned int* __restrict__ dst)
{
    int gid = blockIdx.x * 256 + threadIdx.x;     // 0 .. 20*8192-1
    int c   = gid >> 13;                          // chunk 0..19
    int idx = gid & (CH_DW - 1);
    int k2  = idx >> 9;                           // 0..15
    int n   = idx & 511;                          // coalesced over lanes
    const float* src = (c < 4) ? W_down : W_drift;
    int kc = (c < 4) ? c : (c - 4);
    const float* p = src + (size_t)(kc * 32 + 2 * k2) * DH + n;
    union { _Float16 f[2]; unsigned int u; } pk;
    pk.f[0] = (_Float16)p[0];                     // K even -> low half (RNE)
    pk.f[1] = (_Float16)p[DH];                    // K odd  -> high half
    dst[(size_t)c * CH_DW + n * 16 + k2] = pk.u;
}

template <bool USE_PRE>
__global__ __launch_bounds__(256) void lde_fused(
    const float* __restrict__ x_seq,   const float* __restrict__ aux_feat,
    const float* __restrict__ levy,    const float* __restrict__ W_down,
    const float* __restrict__ b_down,  const float* __restrict__ W_drift,
    const float* __restrict__ b_drift, const float* __restrict__ W_diff1,
    const float* __restrict__ b_diff1, const float* __restrict__ W_diff2,
    const float* __restrict__ b_diff2, const float* __restrict__ W_gate1,
    const float* __restrict__ b_gate1, const float* __restrict__ W_gate2,
    const float* __restrict__ b_gate2, const unsigned int* __restrict__ Wpre,
    float* __restrict__ out)
{
    __shared__ _Float16     hA[ROWS * HA_STRIDE];        // 66.6 KB: f16 shadow of h
    __shared__ unsigned int wbuf[2][DH * WB_STRIDE];     // 80 KB: staged W chunk
    __shared__ float        dscale[ROWS];
    __shared__ float        red[256];

    const int tid    = threadIdx.x;
    const int lane   = tid & 31;
    const int wv     = tid >> 5;            // 0..7
    const int lanelo = lane & 15;
    const int hs     = lane >> 4;           // half-wave select (0/1)
    const int wrow   = (wv & 3) * 16;       // wave row base within block tile
    const int wcol   = (wv >> 2) * 256;     // wave col base (two 256-wide halves)
    const int row0   = blockIdx.x * ROWS;

    const float dt   = 1.0f / 25.0f;
    const float coef = powf(dt, 1.0f / 1.2f);   // dt^(1/alpha), alpha=1.2

    // Slow path: stage + convert one K=32 chunk of f32 W in-kernel.
    auto stage_cvt = [&](const float* __restrict__ Wsrc, int kc, unsigned int* dst) {
        #pragma unroll 4
        for (int i = 0; i < 32; ++i) {
            int flat = tid + i * 256;
            int k2   = flat >> 9;
            int n    = flat & 511;
            const float* p = Wsrc + (size_t)(kc * 32 + 2 * k2) * DH + n;
            float w0 = p[0];
            float w1 = p[DH];
            union { _Float16 f[2]; unsigned int u; } pk;
            pk.f[0] = (_Float16)w0;
            pk.f[1] = (_Float16)w1;
            dst[n * WB_STRIDE + k2] = pk.u;
        }
    };

    // Fast path: pure 16B-granule copy of a pre-converted chunk into padded LDS.
    auto stage_pre = [&](int chunk, unsigned int* dst) {
        const unsigned int* srcBase = Wpre + (size_t)chunk * CH_DW;
        #pragma unroll
        for (int i = 0; i < 8; ++i) {
            int g   = tid + i * 256;          // granule id, 16B each, coalesced
            int n   = g >> 2;
            int k2g = g & 3;
#if HAVE_ASYNC_LDS
            async_cp16(srcBase + g * 4, dst + n * WB_STRIDE + k2g * 4);
#else
            v4u d = *reinterpret_cast<const v4u*>(srcBase + g * 4);
            *reinterpret_cast<v4u*>(dst + n * WB_STRIDE + k2g * 4) = d;
#endif
        }
    };

    // A fragment (16x32 f16) per ISA layout: lane = M row; low lanes K0..7,16..23.
    auto loadA = [&](int kc) {
        ABFrag a;
        int r = wrow + lanelo;
        const unsigned int* base =
            reinterpret_cast<const unsigned int*>(hA) + r * (HA_STRIDE / 2) + kc * 16 + hs * 4;
        a.u[0] = *reinterpret_cast<const v4u*>(base);      // dwords +0..3
        a.u[1] = *reinterpret_cast<const v4u*>(base + 8);  // dwords +8..11
        return a;
    };

    // B fragment (32x16 f16): lane = N col; low lanes K0..7,16..23, high lanes K8..15,24..31.
    auto loadB = [&](const unsigned int* buf, int t) {
        ABFrag b;
        int n = wcol + t * 16 + lanelo;
        const unsigned int* base = buf + n * WB_STRIDE + hs * 4;
        b.u[0] = *reinterpret_cast<const v4u*>(base);
        b.u[1] = *reinterpret_cast<const v4u*>(base + 8);
        return b;
    };

    // ------------------- prologue: stage x_last into hA[:, 0:128] -------------------
    for (int i = 0; i < 32; ++i) {
        int flat = tid + i * 256;               // 64 rows * 128 cols
        int r = flat >> 7, k = flat & 127;
        float x = x_seq[((size_t)(row0 + r) * T_ + (T_ - 1)) * DIN + k];
        hA[r * HA_STRIDE + k] = (_Float16)x;
    }

    float bD[16], bR[16];
    #pragma unroll
    for (int t = 0; t < 16; ++t) {
        bD[t] = b_down[wcol + t * 16 + lanelo];
        bR[t] = b_drift[wcol + t * 16 + lanelo];
    }

    v8f acc[16];
    #pragma unroll
    for (int t = 0; t < 16; ++t) { v8f z = {}; acc[t] = z; }

    // ------------------- h0 = x_last @ W_down + b_down (K=128, 4 chunks) -----------
    if constexpr (USE_PRE) stage_pre(0, wbuf[0]);
    else                   stage_cvt(W_down, 0, wbuf[0]);
    for (int kc = 0; kc < 4; ++kc) {
#if HAVE_ASYNC_LDS
        if constexpr (USE_PRE) wait_async0();
#endif
        __syncthreads();
        if (kc < 3) {
            if constexpr (USE_PRE) stage_pre(kc + 1, wbuf[(kc + 1) & 1]);
            else                   stage_cvt(W_down, kc + 1, wbuf[(kc + 1) & 1]);
        }
        ABFrag a = loadA(kc);
        const unsigned int* cur = wbuf[kc & 1];
        #pragma unroll
        for (int t = 0; t < 16; ++t) {
            ABFrag b = loadB(cur, t);
            acc[t] = __builtin_amdgcn_wmma_f32_16x16x32_f16(
                false, a.h, false, b.h, (short)0, acc[t], false, false);
        }
    }
    __syncthreads();   // all A reads of x-region complete before overwriting hA

    float h[16][8];    // fp32 master copy of h, WMMA C layout
    #pragma unroll
    for (int t = 0; t < 16; ++t)
        #pragma unroll
        for (int v = 0; v < 8; ++v) {
            h[t][v] = acc[t][v] + bD[t];
            hA[(wrow + hs * 8 + v) * HA_STRIDE + (wcol + t * 16 + lanelo)] = (_Float16)h[t][v];
        }

    // ------------------- gate MLP (one row per thread, rows 0..63) ------------------
    float gm = 0.0f;
    if (tid < 64) {
        const float* ar = aux_feat + (size_t)(row0 + tid) * 10;
        float ain[10];
        #pragma unroll
        for (int i = 0; i < 10; ++i) ain[i] = ar[i];
        float g = b_gate2[0];
        #pragma unroll
        for (int o = 0; o < 16; ++o) {
            float u = b_gate1[o];
            #pragma unroll
            for (int i = 0; i < 10; ++i) u = fmaf(ain[i], W_gate1[i * 16 + o], u);
            u = fmaxf(u, 0.0f);
            g = fmaf(u, W_gate2[o], g);
        }
        gm = 0.1f + 1.4f * (1.0f / (1.0f + expf(-g)));   // SIGMA_MIN + range*sigmoid
    }
    __syncthreads();   // hA now holds h0 (f16) for the diffusion head

    // ------------------- diff_micro: relu(h0 @ W_diff1 + b1) @ W_diff2 + b2 --------
    {
        const int r  = tid >> 2;            // 4 threads per row
        const int j0 = (tid & 3) * 25;      // 25 of the 100 hidden units each
        float aj[25];
        #pragma unroll
        for (int jj = 0; jj < 25; ++jj) aj[jj] = b_diff1[j0 + jj];
        for (int k = 0; k < DH; ++k) {
            float hv = (float)hA[r * HA_STRIDE + k];
            const float* wp = W_diff1 + (size_t)k * MID_ + j0;
            #pragma unroll
            for (int jj = 0; jj < 25; ++jj) aj[jj] = fmaf(hv, wp[jj], aj[jj]);
        }
        float part = 0.0f;
        #pragma unroll
        for (int jj = 0; jj < 25; ++jj) part += fmaxf(aj[jj], 0.0f) * W_diff2[j0 + jj];
        red[tid] = part;
    }
    __syncthreads();
    if (tid < 64) {
        float s = red[4 * tid] + red[4 * tid + 1] + red[4 * tid + 2] + red[4 * tid + 3]
                + b_diff2[0];
        float micro = 1.0f / (1.0f + expf(-s));
        dscale[tid] = gm * micro;
    }
    __syncthreads();

    float dsr[8];                            // per-lane row scales, pre-multiplied by coef
    #pragma unroll
    for (int v = 0; v < 8; ++v) dsr[v] = dscale[wrow + hs * 8 + v] * coef;

    // ------------------- main recurrence: 25 steps of h @ W_drift ------------------
    for (int s = 0; s < STEPS_; ++s) {
        #pragma unroll
        for (int t = 0; t < 16; ++t) { v8f z = {}; acc[t] = z; }

        if constexpr (USE_PRE) stage_pre(4, wbuf[0]);
        else                   stage_cvt(W_drift, 0, wbuf[0]);
        for (int kc = 0; kc < 16; ++kc) {
#if HAVE_ASYNC_LDS
            if constexpr (USE_PRE) wait_async0();       // prev-issued chunk landed
#endif
            __syncthreads();                            // publish chunk kc; hA stable
            if (kc < 15) {
                if constexpr (USE_PRE) stage_pre(4 + kc + 1, wbuf[(kc + 1) & 1]);
                else                   stage_cvt(W_drift, kc + 1, wbuf[(kc + 1) & 1]);
            }
            ABFrag a = loadA(kc);
            const unsigned int* cur = wbuf[kc & 1];
            #pragma unroll
            for (int t = 0; t < 16; ++t) {
                ABFrag b = loadB(cur, t);
                acc[t] = __builtin_amdgcn_wmma_f32_16x16x32_f16(
                    false, a.h, false, b.h, (short)0, acc[t], false, false);
            }
        }
        __syncthreads();   // all waves done reading hA before we overwrite it

        const float* es = levy + (size_t)s * B_ * DH;
        #pragma unroll
        for (int t = 0; t < 16; ++t) {
            const int col = wcol + t * 16 + lanelo;
            const float* ep = es + (size_t)(row0 + wrow + hs * 8) * DH + col;
            #pragma unroll
            for (int v = 0; v < 8; ++v) {
                float e  = ep[(size_t)v * DH];            // coalesced across half-wave
                float c  = acc[t][v] + bR[t];
                float dr = fmaxf(c, 0.0f);
                h[t][v]  = fmaf(dr, dt, fmaf(dsr[v], e, h[t][v]));
                hA[(wrow + hs * 8 + v) * HA_STRIDE + col] = (_Float16)h[t][v];
            }
        }
        // next iteration's kc=0 wait+__syncthreads orders hA writes vs. A-frag reads
    }

    // ------------------- epilogue: store fp32 h ------------------------------------
    #pragma unroll
    for (int t = 0; t < 16; ++t) {
        const int col = wcol + t * 16 + lanelo;
        float* op = out + (size_t)(row0 + wrow + hs * 8) * DH + col;
        #pragma unroll
        for (int v = 0; v < 8; ++v) op[(size_t)v * DH] = h[t][v];
    }
}

extern "C" void kernel_launch(void* const* d_in, const int* in_sizes, int n_in,
                              void* d_out, int out_size, void* d_ws, size_t ws_size,
                              hipStream_t stream) {
    (void)in_sizes; (void)n_in; (void)out_size;
    const size_t needWs = (size_t)N_CHUNKS * CH_DW * sizeof(unsigned int);  // 640 KB
    if (d_ws != nullptr && ws_size >= needWs) {
        prep_w<<<dim3(N_CHUNKS * CH_DW / 256), dim3(256), 0, stream>>>(
            (const float*)d_in[3], (const float*)d_in[5], (unsigned int*)d_ws);
        lde_fused<true><<<dim3(B_ / ROWS), dim3(256), 0, stream>>>(
            (const float*)d_in[0],  (const float*)d_in[1],  (const float*)d_in[2],
            (const float*)d_in[3],  (const float*)d_in[4],  (const float*)d_in[5],
            (const float*)d_in[6],  (const float*)d_in[7],  (const float*)d_in[8],
            (const float*)d_in[9],  (const float*)d_in[10], (const float*)d_in[11],
            (const float*)d_in[12], (const float*)d_in[13], (const float*)d_in[14],
            (const unsigned int*)d_ws, (float*)d_out);
    } else {
        lde_fused<false><<<dim3(B_ / ROWS), dim3(256), 0, stream>>>(
            (const float*)d_in[0],  (const float*)d_in[1],  (const float*)d_in[2],
            (const float*)d_in[3],  (const float*)d_in[4],  (const float*)d_in[5],
            (const float*)d_in[6],  (const float*)d_in[7],  (const float*)d_in[8],
            (const float*)d_in[9],  (const float*)d_in[10], (const float*)d_in[11],
            (const float*)d_in[12], (const float*)d_in[13], (const float*)d_in[14],
            (const unsigned int*)nullptr, (float*)d_out);
    }
}
